// LinearQuantized_24713241821311
// MI455X (gfx1250) — compile-verified, matrix operand order checked by reading, not run on hardware
//
#include <hip/hip_runtime.h>
#include <stdint.h>

typedef int v8i __attribute__((ext_vector_type(8)));

#define M_TOK 8192   // B*S
#define K_DIM 4096
#define N_DIM 4096

// Workspace layout:
//   [0]            : uint32 amax bits (abs-max of x)
//   XQ_OFF         : int8 Xq  [M_TOK * K_DIM]  = 33,554,432 B
//   WQ_OFF         : int8 Wq  [N_DIM * K_DIM]  = 16,777,216 B
static constexpr size_t XQ_OFF = 256;
static constexpr size_t WQ_OFF = XQ_OFF + (size_t)M_TOK * K_DIM;

__global__ void k_init(unsigned* amax) { amax[0] = 0u; }

__global__ __launch_bounds__(256) void k_absmax(const float4* __restrict__ x,
                                                unsigned* __restrict__ amax, int n4) {
    __shared__ float red[256];
    float m = 0.0f;
    for (int i = blockIdx.x * blockDim.x + threadIdx.x; i < n4;
         i += gridDim.x * blockDim.x) {
        float4 v = x[i];
        m = fmaxf(m, fmaxf(fmaxf(fabsf(v.x), fabsf(v.y)),
                           fmaxf(fabsf(v.z), fabsf(v.w))));
    }
    red[threadIdx.x] = m;
    __syncthreads();
    for (int s = 128; s > 0; s >>= 1) {
        if (threadIdx.x < s)
            red[threadIdx.x] = fmaxf(red[threadIdx.x], red[threadIdx.x + s]);
        __syncthreads();
    }
    if (threadIdx.x == 0) atomicMax(amax, __float_as_uint(red[0]));
}

__device__ __forceinline__ int q8(float v, float inv) {
    int q = (int)rintf(v * inv);
    return q < -128 ? -128 : (q > 127 ? 127 : q);
}

__global__ __launch_bounds__(256) void k_quant_x(const float4* __restrict__ x,
                                                 const unsigned* __restrict__ amax_bits,
                                                 uint32_t* __restrict__ xq, int n4) {
    int i = blockIdx.x * blockDim.x + threadIdx.x;
    if (i >= n4) return;
    float amax = __uint_as_float(amax_bits[0]);
    float inv  = (amax > 0.0f) ? (127.0f / amax) : 0.0f;
    float4 v = x[i];
    int q0 = q8(v.x, inv), q1 = q8(v.y, inv), q2 = q8(v.z, inv), q3 = q8(v.w, inv);
    xq[i] = (uint32_t)(q0 & 255) | ((uint32_t)(q1 & 255) << 8) |
            ((uint32_t)(q2 & 255) << 16) | ((uint32_t)(q3 & 255) << 24);
}

__global__ __launch_bounds__(256) void k_pack_w(const int4* __restrict__ w,
                                                uint32_t* __restrict__ wq, int n4) {
    int i = blockIdx.x * blockDim.x + threadIdx.x;
    if (i >= n4) return;
    int4 v = w[i];
    wq[i] = (uint32_t)(v.x & 255) | ((uint32_t)(v.y & 255) << 8) |
            ((uint32_t)(v.z & 255) << 16) | ((uint32_t)(v.w & 255) << 24);
}

// One wave computes a 32x32 output supertile (2x2 of 16x16 WMMA tiles).
// 8 waves (256 threads) per block.
__global__ __launch_bounds__(256) void k_gemm(const int8_t* __restrict__ Aq,
                                              const int8_t* __restrict__ Wq,
                                              const unsigned* __restrict__ amax_bits,
                                              const float* __restrict__ wscale,
                                              const int* __restrict__ qbias,
                                              float* __restrict__ out) {
    const int lane  = threadIdx.x & 31;
    const int wave  = threadIdx.x >> 5;
    const int wtile = blockIdx.x * 8 + wave;
    const int nst   = N_DIM / 32;            // supertiles along N
    const int mt    = wtile / nst;
    const int nt    = wtile % nst;

    const int row  = lane & 15;
    const int half = lane >> 4;              // which K half-chunk this lane holds

    // 8-bit A fragment layout (16x64): lane L holds row L%16; per 16-K group,
    // lanes 0-15 hold the low 8 K, lanes 16-31 the high 8 K, as two dwords.
    // Row-major W[N,K] rows are exactly the column-major B operand columns,
    // so B uses the same per-lane addressing with n = lane%16.
    const int8_t* a0 = Aq + (size_t)(mt * 32 + row) * K_DIM + half * 8;
    const int8_t* a1 = a0 + (size_t)16 * K_DIM;
    const int8_t* b0 = Wq + (size_t)(nt * 32 + row) * K_DIM + half * 8;
    const int8_t* b1 = b0 + (size_t)16 * K_DIM;

    v8i acc00 = {0,0,0,0,0,0,0,0};
    v8i acc01 = {0,0,0,0,0,0,0,0};
    v8i acc10 = {0,0,0,0,0,0,0,0};
    v8i acc11 = {0,0,0,0,0,0,0,0};

    for (int k = 0; k < K_DIM; k += 64) {
        v8i va0, va1, vb0, vb1;
#pragma unroll
        for (int g = 0; g < 4; ++g) {
            uint2 t;
            t = *(const uint2*)(a0 + k + g * 16); va0[2*g] = (int)t.x; va0[2*g+1] = (int)t.y;
            t = *(const uint2*)(a1 + k + g * 16); va1[2*g] = (int)t.x; va1[2*g+1] = (int)t.y;
            t = *(const uint2*)(b0 + k + g * 16); vb0[2*g] = (int)t.x; vb0[2*g+1] = (int)t.y;
            t = *(const uint2*)(b1 + k + g * 16); vb1[2*g] = (int)t.x; vb1[2*g+1] = (int)t.y;
        }
        if (k + 64 < K_DIM) {
            __builtin_prefetch(a0 + k + 64, 0, 3);
            __builtin_prefetch(b0 + k + 64, 0, 3);
        }
        // signed x signed int8 -> int32, 16x16x64 per instruction
        acc00 = __builtin_amdgcn_wmma_i32_16x16x64_iu8(true, va0, true, vb0, acc00, false, false);
        acc01 = __builtin_amdgcn_wmma_i32_16x16x64_iu8(true, va0, true, vb1, acc01, false, false);
        acc10 = __builtin_amdgcn_wmma_i32_16x16x64_iu8(true, va1, true, vb0, acc10, false, false);
        acc11 = __builtin_amdgcn_wmma_i32_16x16x64_iu8(true, va1, true, vb1, acc11, false, false);
    }

    // Epilogue: out = (acc + qbias[n]) * wscale[n] * input_scale
    const float is = __uint_as_float(amax_bits[0]) / 127.0f;
    const int col = lane & 15;
    const int n0 = nt * 32 + col;
    const int n1 = n0 + 16;
    const float f0 = wscale[n0] * is;
    const float f1 = wscale[n1] * is;
    const float g0 = (float)qbias[n0];
    const float g1 = (float)qbias[n1];

    // C/D layout: VGPR r, lanes 0-15 -> M = r, lanes 16-31 -> M = r + 8; N = lane%16.
#pragma unroll
    for (int r = 0; r < 8; ++r) {
        const int m0 = mt * 32 + half * 8 + r;
        const int m1 = m0 + 16;
        out[(size_t)m0 * N_DIM + n0] = ((float)acc00[r] + g0) * f0;
        out[(size_t)m0 * N_DIM + n1] = ((float)acc01[r] + g1) * f1;
        out[(size_t)m1 * N_DIM + n0] = ((float)acc10[r] + g0) * f0;
        out[(size_t)m1 * N_DIM + n1] = ((float)acc11[r] + g1) * f1;
    }
}

extern "C" void kernel_launch(void* const* d_in, const int* in_sizes, int n_in,
                              void* d_out, int out_size, void* d_ws, size_t ws_size,
                              hipStream_t stream) {
    const float* x      = (const float*)d_in[0];   // [B,S,IN] fp32
    const int*   w      = (const int*)d_in[1];     // [OUT,IN] int32 (int8-valued)
    const float* wscale = (const float*)d_in[2];   // [OUT]
    const int*   qbias  = (const int*)d_in[3];     // [OUT]
    float*       out    = (float*)d_out;           // [B,S,OUT] fp32

    uint8_t*  ws   = (uint8_t*)d_ws;
    unsigned* amax = (unsigned*)ws;
    uint32_t* xq   = (uint32_t*)(ws + XQ_OFF);
    uint32_t* wq   = (uint32_t*)(ws + WQ_OFF);

    k_init<<<1, 1, 0, stream>>>(amax);

    const int n4x = M_TOK * K_DIM / 4;   // 8,388,608 float4 groups
    k_absmax<<<2048, 256, 0, stream>>>((const float4*)x, amax, n4x);
    k_quant_x<<<n4x / 256, 256, 0, stream>>>((const float4*)x, amax, xq, n4x);

    const int n4w = N_DIM * K_DIM / 4;   // 4,194,304 int4 groups
    k_pack_w<<<n4w / 256, 256, 0, stream>>>((const int4*)w, wq, n4w);

    const int supertiles = (M_TOK / 32) * (N_DIM / 32);  // 32768
    k_gemm<<<supertiles / 8, 256, 0, stream>>>((const int8_t*)(ws + XQ_OFF),
                                               (const int8_t*)(ws + WQ_OFF),
                                               amax, wscale, qbias, out);
}
